// UNetDecoder_13606456394378
// MI455X (gfx1250) — matrix operand order repaired and checked by reading, not compiled, not run on hardware
//
#include <hip/hip_runtime.h>
#include <hip/hip_bf16.h>

typedef __attribute__((ext_vector_type(16))) _Float16 v16h;
typedef __attribute__((ext_vector_type(8)))  _Float16 v8h;
typedef __attribute__((ext_vector_type(8)))  float    v8f;

// ---------------------------------------------------------------------------
// NCHW fp32 -> NHWC f16 conversion with channel-offset placement
// ---------------------------------------------------------------------------
__global__ void cvt_f32_to_f16_kernel(const float* __restrict__ in,
                                      _Float16* __restrict__ out,
                                      int total, int HW, int C, int Ctot, int c0)
{
    int idx = blockIdx.x * blockDim.x + threadIdx.x;
    if (idx >= total) return;
    int pix = idx % HW;
    int c   = (idx / HW) % C;
    int b   = idx / (HW * C);
    size_t oidx = ((size_t)b * HW + pix) * (size_t)Ctot + (c0 + c);
    out[oidx] = (_Float16)in[idx];
}

// ---------------------------------------------------------------------------
// weight reorder + cvt: OIHW fp32 -> f16 [O][tap-major K, natural channel order]
//   slot = kloc: channels keep natural order within each 32-channel block
// lane half h then reads its 16 A-fragment values contiguously at offset h*16.
// ---------------------------------------------------------------------------
__global__ void wreorder_kernel(const float* __restrict__ in,
                                _Float16* __restrict__ out,
                                int total, int Cin)
{
    int idx = blockIdx.x * blockDim.x + threadIdx.x;
    if (idx >= total) return;
    int K  = Cin * 9;
    int o  = idx / K;
    int r  = idx - o * K;
    int c  = r / 9;
    int t9 = r - c * 9;
    int cb   = c & ~31;
    int kloc = c & 31;
    int slot = kloc;
    out[(size_t)o * K + (size_t)t9 * Cin + cb + slot] = (_Float16)in[idx];
}

// ---------------------------------------------------------------------------
// zero the 4KB padding page (read target for out-of-bounds conv taps)
// ---------------------------------------------------------------------------
__global__ void zerofill_kernel(unsigned* p)
{
    uint4 z = {0u, 0u, 0u, 0u};
    ((uint4*)p)[threadIdx.x] = z;   // 256 * 16B = 4KB
}

// ---------------------------------------------------------------------------
// style modulation: s[b][o] = dot(style[b,:256], fc_w[o,:256]) + fc_b[o]
// ---------------------------------------------------------------------------
__global__ void style_kernel(const float* __restrict__ style,
                             const float* __restrict__ fc_w,
                             const float* __restrict__ fc_b,
                             float* __restrict__ s,
                             int Cout)
{
    __shared__ float red[256];
    int bo = blockIdx.x;
    int b = bo / Cout, o = bo % Cout;
    int t = threadIdx.x;
    red[t] = style[b * 256 + t] * fc_w[o * 256 + t];
    __syncthreads();
    for (int off = 128; off > 0; off >>= 1) {
        if (t < off) red[t] += red[t + off];
        __syncthreads();
    }
    if (t == 0) s[bo] = red[0] + fc_b[o];
}

// ---------------------------------------------------------------------------
// Fused implicit-GEMM 3x3 conv (stride 1, pad 1), optional nearest-2x upsample
// folded into addressing.  NHWC f16 activations.
//
// B tile staged with GLOBAL_LOAD_ASYNC_TO_LDS_B128 (ASYNCcnt DMA): each thread
// moves one 64B row with a 4-instruction async clause; padding lanes redirect
// their per-lane offset to a zeroed 4KB page (no divergent branch, no EXEC
// juggling).  A fragments are single v16h loads from the tap-major
// L2-resident weight matrix.  B fragments are single v16h LDS loads.
//
// Block tile 64(M) x 256(N); 8 waves as 4(M) x 2(N); each wave 16x128
// -> 8 x v_wmma_f32_16x16x32_f16 per (tap, 32-channel block) from 1 A frag.
// ---------------------------------------------------------------------------
template <int CIN, int HIN, int WIN, int UP, int COUT, int RELU,
          int CTOT, int HAS_SCALE, int HAS_BIAS, int OUT32>
__global__ __launch_bounds__(256)
void conv_wmma_kernel(const char* __restrict__ base,    // d_ws base
                      unsigned inOff,                   // byte off of input
                      unsigned zOff,                    // byte off of zero page
                      const _Float16* __restrict__ wgt, // [COUT,KTOT] f16
                      const float* __restrict__ scale,  // [8,COUT] or null
                      const float* __restrict__ bias,   // [COUT] or null
                      _Float16* __restrict__ out16,     // [8,HOUT,WOUT,CTOT]
                      float* __restrict__ out32)        // [8,COUT,HOUT,WOUT]
{
    constexpr int HOUT = HIN << UP;
    constexpr int WOUT = WIN << UP;
    constexpr int HW   = HOUT * WOUT;
    constexpr int KTOT = CIN * 9;

    __shared__ __align__(32) _Float16 Btile[256 * 32];

    const int tid   = threadIdx.x;
    const int wave  = tid >> 5;
    const int lane  = tid & 31;
    const int mWave = wave & 3;    // 0..3 -> M sub-block of 16
    const int nWave = wave >> 2;   // 0..1 -> N sub-block of 128

    const int mTileBase = blockIdx.y * 64;
    const int nTileBase = blockIdx.x * 256;

    const int half = lane >> 4;
    const int l16  = lane & 15;

    // direct A-fragment row (clamped; rows >= COUT are discarded at store)
    int mA = mTileBase + mWave * 16 + l16;
    if (mA >= COUT) mA = COUT - 1;
    const size_t aRow = (size_t)mA * KTOT + half * 16;

    // B staging: 256 rows x 32 k; each thread stages one 64B row
    _Float16* ldsDst = Btile + tid * 32;
    const unsigned ldsAddr = (unsigned)(uintptr_t)ldsDst;   // LDS byte offset

    // decode the output pixel this thread stages (compile-time shifts)
    const int nG = nTileBase + tid;
    const int bB = nG / HW;
    const int remB = nG % HW;
    const int yB = remB / WOUT;
    const int xB = remB % WOUT;

    v8f acc[8] = {};

    #pragma unroll
    for (int t9 = 0; t9 < 9; ++t9) {
        const int dy = t9 / 3;          // literal after unroll
        const int dx = t9 - dy * 3;

        // per-tap source pixel; padding lanes redirect to the zero page
        const int iy = yB + dy - 1;
        const int ix = xB + dx - 1;
        const bool valid = (iy >= 0) & (iy < HOUT) & (ix >= 0) & (ix < WOUT);
        const int sy = (iy >= 0 ? iy : 0) >> UP;
        const int sx = (ix >= 0 ? ix : 0) >> UP;
        const size_t pixBase = (((size_t)bB * HIN + sy) * WIN + sx) * CIN;
        const unsigned voffBase =
            valid ? (inOff + (unsigned)(pixBase * 2)) : zOff;
        __builtin_prefetch(base + voffBase, 0, 1);   // global_prefetch_b8

        for (int cb = 0; cb < CIN; cb += 32) {
            __syncthreads();
            // ---- stage B tile: 64B/thread async DMA global->LDS ----
            {
                unsigned voff = voffBase + (unsigned)(cb * 2);
                asm volatile(
                    "global_load_async_to_lds_b128 %0, %1, %2\n\t"
                    "global_load_async_to_lds_b128 %0, %1, %2 offset:16\n\t"
                    "global_load_async_to_lds_b128 %0, %1, %2 offset:32\n\t"
                    "global_load_async_to_lds_b128 %0, %1, %2 offset:48"
                    :: "v"(ldsAddr), "v"(voff),
                       "s"((unsigned long long)(uintptr_t)base)
                    : "memory");
            }
            asm volatile("s_wait_asynccnt 0x0" ::: "memory");
            __syncthreads();

            // ---- A fragment: one v16h from tap-major weights (global) ----
            v16h aFrag = *(const v16h*)(wgt + aRow + t9 * CIN + cb);

            #pragma unroll
            for (int i = 0; i < 8; ++i) {
                // B fragment: contiguous 32B in LDS (lane half -> K 0..15/16..31)
                const _Float16* Br =
                    Btile + (nWave * 128 + i * 16 + l16) * 32 + half * 16;
                v16h bFrag = *(const v16h*)Br;
                acc[i] = __builtin_amdgcn_wmma_f32_16x16x32_f16(
                    false, aFrag, false, bFrag, (short)0, acc[i], false, false);
            }
        }
    }

    // ---- epilogue: scale / bias / relu ----
    const int mBase8 = mTileBase + mWave * 16 + 8 * half;  // 8 consecutive ch
    #pragma unroll
    for (int i = 0; i < 8; ++i) {
        const int n = nTileBase + nWave * 128 + i * 16 + l16;  // global pixel
        const int b = n / HW;
        if constexpr (OUT32) {
            // NCHW f32 output (final 3-channel stage): scatter with guard
            int rem = n % HW;
            #pragma unroll
            for (int r = 0; r < 8; ++r) {
                int m = mBase8 + r;
                if (m >= COUT) continue;
                float o = acc[i][r];
                if constexpr (HAS_SCALE) o *= scale[b * COUT + m];
                if constexpr (HAS_BIAS)  o += bias[m];
                if constexpr (RELU)      o = o < 0.0f ? 0.0f : o;
                out32[((size_t)b * COUT + m) * HW + rem] = o;
            }
        } else {
            // NHWC f16 output: one 16B vector store of 8 channels
            v8h hv;
            #pragma unroll
            for (int r = 0; r < 8; ++r) {
                int m = mBase8 + r;
                float o = acc[i][r];
                if constexpr (HAS_SCALE) o *= scale[b * COUT + m];
                if constexpr (HAS_BIAS)  o += bias[m];
                if constexpr (RELU)      o = o < 0.0f ? 0.0f : o;
                hv[r] = (_Float16)o;
            }
            *(v8h*)(out16 + (size_t)n * CTOT + mBase8) = hv;
        }
    }
}

// ---------------------------------------------------------------------------
extern "C" void kernel_launch(void* const* d_in, const int* in_sizes, int n_in,
                              void* d_out, int out_size, void* d_ws, size_t ws_size,
                              hipStream_t stream) {
    (void)in_sizes; (void)n_in; (void)out_size; (void)ws_size;

    const float* x     = (const float*)d_in[0];   // [8,256,64,64]
    const float* enc1  = (const float*)d_in[1];   // [8,64,256,256]
    const float* enc2  = (const float*)d_in[2];   // [8,128,128,128]
    const float* enc3  = (const float*)d_in[3];   // [8,256,64,64]
    const float* style = (const float*)d_in[4];   // [8,256]
    const float* w1    = (const float*)d_in[5];   // [128,512,3,3]
    const float* fcw1  = (const float*)d_in[6];   // [128,256]
    const float* fcb1  = (const float*)d_in[7];   // [128]
    const float* w2    = (const float*)d_in[8];   // [64,256,3,3]
    const float* fcw2  = (const float*)d_in[9];   // [64,256]
    const float* fcb2  = (const float*)d_in[10];  // [64]
    const float* fw1   = (const float*)d_in[11];  // [64,128,3,3]
    const float* fb1   = (const float*)d_in[12];  // [64]
    const float* fw2   = (const float*)d_in[13];  // [3,64,3,3]
    const float* fb2   = (const float*)d_in[14];  // [3]

    // ---- workspace carve-up (256B aligned) ----
    char*  ws  = (char*)d_ws;
    size_t off = 0;
    auto carve = [&](size_t bytes) -> void* {
        void* p = ws + off;
        off += (bytes + 255) & ~(size_t)255;
        return p;
    };
    // NHWC f16 activation buffers
    _Float16* concat0 = (_Float16*)carve((size_t)8 * 64 * 64 * 512 * 2);
    _Float16* concat1 = (_Float16*)carve((size_t)8 * 128 * 128 * 256 * 2);
    _Float16* concat2 = (_Float16*)carve((size_t)8 * 256 * 256 * 128 * 2);
    _Float16* h3buf   = (_Float16*)carve((size_t)8 * 256 * 256 * 64 * 2);
    _Float16* wh1     = (_Float16*)carve((size_t)128 * 4608 * 2);
    _Float16* wh2     = (_Float16*)carve((size_t)64 * 2304 * 2);
    _Float16* wh3     = (_Float16*)carve((size_t)64 * 1152 * 2);
    _Float16* wh4     = (_Float16*)carve((size_t)3 * 576 * 2);
    float*    s1      = (float*)carve((size_t)8 * 128 * 4);
    float*    s2      = (float*)carve((size_t)8 * 64 * 4);
    unsigned* zpage   = (unsigned*)carve(4096);   // zero page for padding taps

    auto woff = [&](const void* p) -> unsigned {
        return (unsigned)((const char*)p - ws);
    };

    auto cvt = [&](const float* src, _Float16* dst, int total,
                   int HW, int C, int Ctot, int c0) {
        cvt_f32_to_f16_kernel<<<(total + 255) / 256, 256, 0, stream>>>(
            src, dst, total, HW, C, Ctot, c0);
    };
    auto wre = [&](const float* src, _Float16* dst, int Cout, int Cin) {
        int total = Cout * Cin * 9;
        wreorder_kernel<<<(total + 255) / 256, 256, 0, stream>>>(
            src, dst, total, Cin);
    };

    // ---- stage 0: zero page, f16 conversion, NCHW->NHWC, concats ----
    zerofill_kernel<<<1, 256, 0, stream>>>(zpage);
    cvt(x,    concat0, 8 * 256 * 64 * 64,   4096,  256, 512, 0);
    cvt(enc3, concat0, 8 * 256 * 64 * 64,   4096,  256, 512, 256);
    cvt(enc2, concat1, 8 * 128 * 128 * 128, 16384, 128, 256, 128);
    cvt(enc1, concat2, 8 * 64 * 256 * 256,  65536, 64,  128, 64);
    wre(w1,  wh1, 128, 512);
    wre(w2,  wh2, 64,  256);
    wre(fw1, wh3, 64,  128);
    wre(fw2, wh4, 3,   64);

    // ---- style modulation scales ----
    style_kernel<<<8 * 128, 256, 0, stream>>>(style, fcw1, fcb1, s1, 128);
    style_kernel<<<8 * 64,  256, 0, stream>>>(style, fcw2, fcb2, s2, 64);

    const unsigned zOff = woff(zpage);

    // ---- conv1: concat0[8,64,64,512] --up2--> modconv 512->128 @128x128 ----
    conv_wmma_kernel<512, 64, 64, 1, 128, 1, 256, 1, 0, 0>
        <<<dim3(8 * 128 * 128 / 256, 2), 256, 0, stream>>>(
            ws, woff(concat0), zOff, wh1, s1, nullptr, concat1, nullptr);

    // ---- conv2: concat1[8,128,128,256] --up2--> modconv 256->64 @256x256 ----
    conv_wmma_kernel<256, 128, 128, 1, 64, 1, 128, 1, 0, 0>
        <<<dim3(8 * 256 * 256 / 256, 1), 256, 0, stream>>>(
            ws, woff(concat1), zOff, wh2, s2, nullptr, concat2, nullptr);

    // ---- conv3: concat2[8,256,256,128] -> 64ch + bias + relu ----
    conv_wmma_kernel<128, 256, 256, 0, 64, 1, 64, 0, 1, 0>
        <<<dim3(8 * 256 * 256 / 256, 1), 256, 0, stream>>>(
            ws, woff(concat2), zOff, wh3, nullptr, fb1, h3buf, nullptr);

    // ---- conv4: h3[8,256,256,64] -> 3ch + bias (f32 NCHW out) ----
    conv_wmma_kernel<64, 256, 256, 0, 3, 0, 3, 0, 1, 1>
        <<<dim3(8 * 256 * 256 / 256, 1), 256, 0, stream>>>(
            ws, woff(h3buf), zOff, wh4, nullptr, fb2, nullptr, (float*)d_out);
}
